// Attention_15771119911478
// MI455X (gfx1250) — matrix operand order repaired and checked by reading, not compiled, not run on hardware
//
#include <hip/hip_runtime.h>
#include <hip/hip_bf16.h>

#define BATCH   2
#define SEQ     2048
#define DHEAD   64
#define NHEADS  8
#define HD      (NHEADS * DHEAD)   // 512

typedef __bf16 bf16;
typedef __bf16 v16bf __attribute__((ext_vector_type(16)));
typedef __bf16 v8bf  __attribute__((ext_vector_type(8)));
typedef __bf16 v4bf  __attribute__((ext_vector_type(4)));
typedef float  v8f   __attribute__((ext_vector_type(8)));
typedef float  f32x4 __attribute__((ext_vector_type(4)));
typedef int    v4i   __attribute__((ext_vector_type(4)));

// pointee types for the async-copy builtin: int4 in addrspace(1)/(3)
typedef __attribute__((address_space(1))) v4i gv4i;
typedef __attribute__((address_space(3))) v4i lv4i;

// -------- async global->LDS copy (gfx1250), compile-safe via __has_builtin ---
#if defined(__has_builtin)
#  if __has_builtin(__builtin_amdgcn_global_load_async_to_lds_b128)
#    define HAVE_ASYNC_LDS 1
#  endif
#  if __has_builtin(__builtin_amdgcn_s_wait_asynccnt)
#    define HAVE_WAIT_ASYNC 1
#  endif
#endif
#ifndef HAVE_ASYNC_LDS
#  define HAVE_ASYNC_LDS 0
#endif
#ifndef HAVE_WAIT_ASYNC
#  define HAVE_WAIT_ASYNC 0
#endif

static __device__ __forceinline__ void async_copy16(const bf16* g, bf16* l) {
#if HAVE_ASYNC_LDS
  __builtin_amdgcn_global_load_async_to_lds_b128((gv4i*)g, (lv4i*)l, 0, 0);
#else
  *(v8bf*)l = *(const v8bf*)g;          // synchronous fallback
#endif
}

static __device__ __forceinline__ void async_wait0() {
#if HAVE_ASYNC_LDS
#  if HAVE_WAIT_ASYNC
  __builtin_amdgcn_s_wait_asynccnt(0);
#  else
  asm volatile("s_wait_asynccnt 0" ::: "memory");
#  endif
#endif
}

static __device__ __forceinline__ v8f wmma_bf16f32(v16bf a, v16bf b, v8f c) {
  // D = A(16x32 bf16) * B(32x16 bf16) + C(16x16 f32)
  return __builtin_amdgcn_wmma_f32_16x16x32_bf16(false, a, false, b, (short)0, c,
                                                 false, false);
}

// ---------------------------------------------------------------------------
// Kernel 0: one-time f32 -> bf16 conversion of q and all weights.
// ---------------------------------------------------------------------------
__global__ void __launch_bounds__(256)
cvt_kernel(const float* __restrict__ q,  const float* __restrict__ Wq,
           const float* __restrict__ Wk, const float* __restrict__ Wv,
           const float* __restrict__ Wo,
           bf16* __restrict__ qb,  bf16* __restrict__ Wqb,
           bf16* __restrict__ Wkb, bf16* __restrict__ Wvb,
           bf16* __restrict__ Wob)
{
  const int NQ = BATCH * SEQ * DHEAD / 4;   // 65536 vec4 units
  const int NW = HD * DHEAD / 4;            // 8192 vec4 units
  int idx = blockIdx.x * 256 + threadIdx.x; // 98304 total
  const float* src;
  bf16* dst;
  if (idx < NQ)              { src = q;  dst = qb;  }
  else if ((idx -= NQ) < NW) { src = Wq; dst = Wqb; }
  else if ((idx -= NW) < NW) { src = Wk; dst = Wkb; }
  else if ((idx -= NW) < NW) { src = Wv; dst = Wvb; }
  else      { idx -= NW;       src = Wo; dst = Wob; }
  f32x4 v = *(const f32x4*)(src + (size_t)idx * 4);
  v4bf o;
#pragma unroll
  for (int i = 0; i < 4; ++i) o[i] = (bf16)v[i];
  *(v4bf*)(dst + (size_t)idx * 4) = o;
}

// ---------------------------------------------------------------------------
// Kernel 1: fused QKV projection + RoPE + row norms.
// ---------------------------------------------------------------------------
__global__ void __launch_bounds__(128)
proj_rope_kernel(const bf16* __restrict__ qbf,
                 const bf16* __restrict__ Wqb,
                 const bf16* __restrict__ Wkb,
                 const bf16* __restrict__ Wvb,
                 bf16* __restrict__ Qb, bf16* __restrict__ Kb,
                 bf16* __restrict__ Vb,
                 float* __restrict__ Qn, float* __restrict__ Kn)
{
  __shared__ float norm_s[4][16];

  const int stile = blockIdx.x;
  const int b     = blockIdx.y;
  const int mh    = blockIdx.z;
  const int mat   = mh >> 3;          // 0=q 1=k 2=v (uniform per block)
  const int h     = mh & 7;
  const int wave  = threadIdx.x >> 5;
  const int lane  = threadIdx.x & 31;
  const int ln    = lane & 15;
  const int hi    = lane >> 4;
  const int s0    = stile * 16;
  const int ka    = hi * 16;
  const int bh    = b * NHEADS + h;

  const bf16* __restrict__ W = (mat == 0) ? Wqb : (mat == 1) ? Wkb : Wvb;

  const bf16* qrow = qbf + (size_t)(b * SEQ + s0 + ln) * DHEAD;
  v16bf a0, a1;
  {
    v8bf t0 = *(const v8bf*)(qrow + ka);
    v8bf t1 = *(const v8bf*)(qrow + 8 + ka);
    v8bf t2 = *(const v8bf*)(qrow + 32 + ka);
    v8bf t3 = *(const v8bf*)(qrow + 40 + ka);
#pragma unroll
    for (int i = 0; i < 8; ++i) {
      a0[i] = t0[i]; a0[8 + i] = t1[i];
      a1[i] = t2[i]; a1[8 + i] = t3[i];
    }
  }

  const int dh = wave * 16 + ln;         // dim within head, 0..63
  const int dg = h * DHEAD + dh;         // row of W
  const bf16* wrow = W + (size_t)dg * DHEAD + hi * 16;
  v16bf b0 = *(const v16bf*)(wrow);
  v16bf b1 = *(const v16bf*)(wrow + 32);

  v8f c = {};
  c = wmma_bf16f32(a0, b0, c);
  c = wmma_bf16f32(a1, b1, c);

  if (mat < 2) {
    const float inv = __expf(-(float)(dh & ~1) * (9.210340371976184f / 64.0f));
    const bool even = ((dh & 1) == 0);
#pragma unroll
    for (int j = 0; j < 8; ++j) {
      float sn, cs;
      __sincosf((float)(s0 + j + 8 * hi) * inv, &sn, &cs);
      float x = c[j];
      float w = __shfl_xor(x, 1, 32);
      c[j] = even ? (x * cs - w * sn) : (w * sn + x * cs);
    }

#pragma unroll
    for (int j = 0; j < 8; ++j) {
      float s = c[j] * c[j];
      s += __shfl_xor(s, 1, 32);
      s += __shfl_xor(s, 2, 32);
      s += __shfl_xor(s, 4, 32);
      s += __shfl_xor(s, 8, 32);
      if (ln == 0) norm_s[wave][j + 8 * hi] = s;   // lanes 0 & 16
    }
    __syncthreads();                               // uniform (mat per-block)
    if (threadIdx.x < 16) {
      float t = norm_s[0][threadIdx.x] + norm_s[1][threadIdx.x] +
                norm_s[2][threadIdx.x] + norm_s[3][threadIdx.x];
      float* __restrict__ N = (mat == 0) ? Qn : Kn;
      N[(size_t)bh * SEQ + s0 + threadIdx.x] = t;
    }
  }

  if (mat == 2) {
    bf16* out = Vb + ((size_t)bh * DHEAD + dh) * SEQ + s0;
#pragma unroll
    for (int j = 0; j < 8; ++j) out[j + 8 * hi] = (bf16)c[j];
  } else {
    bf16* out = ((mat == 0) ? Qb : Kb) + ((size_t)bh * SEQ + s0) * DHEAD + dh;
#pragma unroll
    for (int j = 0; j < 8; ++j) out[(size_t)(j + 8 * hi) * DHEAD] = (bf16)c[j];
  }
}

// ---------------------------------------------------------------------------
// Kernel 2: streaming causal distance-attention with double-buffered async
// global->LDS K/V staging (ASYNCcnt pipeline).
// ---------------------------------------------------------------------------
__global__ void __launch_bounds__(128)
attn_kernel(const bf16* __restrict__ Qb, const bf16* __restrict__ Kb,
            const bf16* __restrict__ Vb, const float* __restrict__ gamma,
            const float* __restrict__ Qn, const float* __restrict__ Kn,
            bf16* __restrict__ AOb)
{
  __shared__ __align__(32) bf16 ks[2][32 * DHEAD];   // 2 x 4KB K tiles
  __shared__ __align__(32) bf16 vs[2][DHEAD * 32];   // 2 x 4KB V tiles
  __shared__ __align__(32) bf16 sc_s[4][16][32];     // per-wave score tiles

  const int bh   = blockIdx.y;
  const int b    = bh >> 3;
  const int h    = bh & 7;
  const int tid  = threadIdx.x;
  const int wave = tid >> 5;
  const int lane = tid & 31;
  const int ln   = lane & 15;
  const int hi   = lane >> 4;
  const int sb   = blockIdx.x * 64;
  const int r0   = sb + wave * 16;
  const float coef = -gamma[h] * 0.125f;             // -gamma / sqrt(64)
  const int ka = hi * 16;

  // Q tile in A layout
  const bf16* qrow = Qb + ((size_t)bh * SEQ + r0 + ln) * DHEAD;
  v16bf a0, a1;
  {
    v8bf t0 = *(const v8bf*)(qrow + ka);
    v8bf t1 = *(const v8bf*)(qrow + 8 + ka);
    v8bf t2 = *(const v8bf*)(qrow + 32 + ka);
    v8bf t3 = *(const v8bf*)(qrow + 40 + ka);
#pragma unroll
    for (int i = 0; i < 8; ++i) {
      a0[i] = t0[i]; a0[8 + i] = t1[i];
      a1[i] = t2[i]; a1[8 + i] = t3[i];
    }
  }

  // loop-invariant query norms for this wave's 8 rows (per half-wave)
  const float* qnp = Qn + (size_t)bh * SEQ + r0 + 8 * hi;
  const f32x4 qn0 = *(const f32x4*)(qnp);
  const f32x4 qn1 = *(const f32x4*)(qnp + 4);
  const float* knp = Kn + (size_t)bh * SEQ;

  v8f o[4] = {{}, {}, {}, {}};

  const int ntiles = (sb >> 5) + 2;                  // (sb+64)/32

  // cooperative async stage of tile `it` into buffer it&1
  auto stage = [&](int it) {
    const int tb = it * 32;
    bf16* kd = &ks[it & 1][0];
    bf16* vd = &vs[it & 1][0];
    const bf16* ksrc = Kb + ((size_t)bh * SEQ + tb) * DHEAD;  // 4KB contiguous
#pragma unroll
    for (int p = 0; p < 2; ++p) {
      const int off = tid * 8 + p * 1024;
      async_copy16(ksrc + off, kd + off);
    }
#pragma unroll
    for (int p = 0; p < 2; ++p) {
      const int c   = tid + p * 128;                 // 256 16B chunks
      const int row = c >> 2;                        // dim row 0..63
      const int prt = c & 3;
      async_copy16(Vb + ((size_t)bh * DHEAD + row) * SEQ + tb + prt * 8,
                   vd + row * 32 + prt * 8);
    }
  };

  stage(0);

  for (int it = 0; it < ntiles; ++it) {
    const int tbase = it * 32;
    async_wait0();                                   // own staged chunks done
    __syncthreads();                                 // whole tile visible
    if (it + 1 < ntiles) stage(it + 1);              // overlap next tile DMA

    if (tbase <= r0 + 15) {                          // causal range of wave
      const bool full = (tbase + 31 <= r0);
      const bf16* kbuf = &ks[it & 1][0];
      const bf16* vbuf = &vs[it & 1][0];

#pragma unroll
      for (int kt = 0; kt < 2; ++kt) {
        const int tb = tbase + kt * 16;
        const bf16* krow = kbuf + (kt * 16 + ln) * DHEAD + hi * 16;
        v16bf kb0 = *(const v16bf*)(krow);
        v16bf kb1 = *(const v16bf*)(krow + 32);
        const float kn = knp[tb + ln];               // per key column

        v8f c = {};
        c = wmma_bf16f32(a0, kb0, c);
        c = wmma_bf16f32(a1, kb1, c);

#pragma unroll
        for (int j = 0; j < 8; ++j) {
          const float qn_j = (j < 4) ? qn0[j] : qn1[j - 4];
          float dist = fmaxf(qn_j - 2.0f * c[j] + kn, 0.0f);
          float sc = __expf(coef * dist);
          if (!full) {
            const int sg = r0 + j + 8 * hi;          // query row
            const int tg = tb + ln;                  // key column
            if (tg > sg) sc = 0.0f;
          }
          sc_s[wave][j + 8 * hi][kt * 16 + ln] = (bf16)sc;
        }
      }

      // LDS write -> read re-layout hazard (same wave; DS is in-order)
      asm volatile("s_wait_dscnt 0" ::: "memory");

      v16bf sA;                                      // scores as A fragment
      {
        v8bf u0 = *(const v8bf*)(&sc_s[wave][ln][ka]);
        v8bf u1 = *(const v8bf*)(&sc_s[wave][ln][8 + ka]);
#pragma unroll
        for (int i = 0; i < 8; ++i) { sA[i] = u0[i]; sA[8 + i] = u1[i]; }
      }

#pragma unroll
      for (int n = 0; n < 4; ++n) {
        v16bf vb = *(const v16bf*)(vbuf + (n * 16 + ln) * 32 + hi * 16);
        o[n] = wmma_bf16f32(sA, vb, o[n]);
      }
    }
  }

  // store attention output, layout [B, S, H*D] bf16
  bf16* ob = AOb + (size_t)(b * SEQ + r0) * HD + h * DHEAD;
#pragma unroll
  for (int n = 0; n < 4; ++n)
#pragma unroll
    for (int j = 0; j < 8; ++j)
      ob[(size_t)(j + 8 * hi) * HD + n * 16 + ln] = (bf16)o[n][j];
}

// ---------------------------------------------------------------------------
// Kernel 3: output projection, (B*S x 512) @ Wo^T -> (B*S x 64) f32.
// ---------------------------------------------------------------------------
__global__ void __launch_bounds__(128)
oproj_kernel(const bf16* __restrict__ AOb, const bf16* __restrict__ Wob,
             float* __restrict__ out)
{
  const int row0 = blockIdx.x * 16;
  const int wave = threadIdx.x >> 5;
  const int lane = threadIdx.x & 31;
  const int ln = lane & 15;
  const int hi = lane >> 4;
  const int ka = hi * 16;

  const bf16* arow = AOb + (size_t)(row0 + ln) * HD;
  const bf16* wrow = Wob + (size_t)(wave * 16 + ln) * HD + hi * 16;

  v8f c = {};
#pragma unroll 4
  for (int kc = 0; kc < 16; ++kc) {
    v16bf a;
    {
      v8bf u0 = *(const v8bf*)(arow + kc * 32 + ka);
      v8bf u1 = *(const v8bf*)(arow + kc * 32 + 8 + ka);
#pragma unroll
      for (int i = 0; i < 8; ++i) { a[i] = u0[i]; a[8 + i] = u1[i]; }
    }
    v16bf bb = *(const v16bf*)(wrow + kc * 32);
    c = wmma_bf16f32(a, bb, c);
  }
#pragma unroll
  for (int j = 0; j < 8; ++j)
    out[(size_t)(row0 + j + 8 * hi) * DHEAD + wave * 16 + ln] = c[j];
}

// ---------------------------------------------------------------------------
extern "C" void kernel_launch(void* const* d_in, const int* in_sizes, int n_in,
                              void* d_out, int out_size, void* d_ws,
                              size_t ws_size, hipStream_t stream) {
  (void)in_sizes; (void)n_in; (void)out_size; (void)ws_size;
  const float* q     = (const float*)d_in[0];
  const float* Wq    = (const float*)d_in[1];
  const float* Wk    = (const float*)d_in[2];
  const float* Wv    = (const float*)d_in[3];
  const float* Wo    = (const float*)d_in[4];
  const float* gamma = (const float*)d_in[5];

  const size_t NE = (size_t)BATCH * NHEADS * SEQ * DHEAD;  // 2,097,152
  const size_t NQ = (size_t)BATCH * SEQ * DHEAD;           // 262,144
  const size_t NW = (size_t)HD * DHEAD;                    // 32,768
  bf16* Qb  = (bf16*)d_ws;
  bf16* Kb  = Qb + NE;
  bf16* Vb  = Kb + NE;
  bf16* AOb = Vb + NE;
  bf16* qbf = AOb + NE;
  bf16* Wqb = qbf + NQ;
  bf16* Wkb = Wqb + NW;
  bf16* Wvb = Wkb + NW;
  bf16* Wob = Wvb + NW;
  float* Qn = (float*)(Wob + NW);                          // 4B-aligned
  float* Kn = Qn + (size_t)BATCH * NHEADS * SEQ;           // ~17.8 MB total

  cvt_kernel<<<dim3(384), 256, 0, stream>>>(q, Wq, Wk, Wv, Wo,
                                            qbf, Wqb, Wkb, Wvb, Wob);
  proj_rope_kernel<<<dim3(SEQ / 16, BATCH, 3 * NHEADS), 128, 0, stream>>>(
      qbf, Wqb, Wkb, Wvb, Qb, Kb, Vb, Qn, Kn);
  attn_kernel<<<dim3(SEQ / 64, BATCH * NHEADS), 128, 0, stream>>>(
      Qb, Kb, Vb, gamma, Qn, Kn, AOb);
  oproj_kernel<<<dim3((BATCH * SEQ) / 16), 128, 0, stream>>>(AOb, Wob,
                                                             (float*)d_out);
}
